// HyenaAttention_34308198761144
// MI455X (gfx1250) — matrix operand order repaired and checked by reading, not compiled, not run on hardware
//
#include <hip/hip_runtime.h>
#include <hip/hip_bf16.h>
#include <math.h>

#define DIMC  512
#define HIDC  128
#define OUTC  32
#define ORD   3
#define BATCH 4
#define SEQ   4096
#define NPROJ 2048   // DIM*(ORDER+1)

typedef __bf16 bf16;
typedef __attribute__((ext_vector_type(16))) __bf16 v16bf;
typedef __attribute__((ext_vector_type(8)))  __bf16 v8bf;
typedef __attribute__((ext_vector_type(8)))  float  v8f;

// ---------------- WMMA fragment loaders (wave32, v_wmma_f32_16x16x32_bf16) ----
// A tile in LDS: row-major [rows][32] bf16. Lane l<16 -> M=row0+l, K blocks {0..7,16..23};
// lanes 16..31 -> M=row0+l-16, K blocks {8..15,24..31}.
__device__ __forceinline__ v16bf frag_a(const bf16* tile, int row0, int lane) {
  const bf16* base = tile + (row0 + (lane & 15)) * 32;
  int kb = (lane >> 4) * 8;
  v8bf lo = *(const v8bf*)(base + kb);
  v8bf hi = *(const v8bf*)(base + kb + 16);
  v16bf r;
#pragma unroll
  for (int i = 0; i < 8; ++i) { r[i] = lo[i]; r[i + 8] = hi[i]; }
  return r;
}
// B tile staged transposed in LDS: [n][32] bf16 (K contiguous per column).
__device__ __forceinline__ v16bf frag_b(const bf16* tileT, int n0, int lane) {
  const bf16* base = tileT + (n0 + (lane & 15)) * 32 + (lane >> 4) * 8;
  v8bf lo = *(const v8bf*)(base);
  v8bf hi = *(const v8bf*)(base + 16);
  v16bf r;
#pragma unroll
  for (int i = 0; i < 8; ++i) { r[i] = lo[i]; r[i + 8] = hi[i]; }
  return r;
}

// ---------------- GEMM1: hat_z(bf16) = u @ pW + pb   (16384x512x2048) --------
// Double-buffered LDS, register staging, one barrier per K-step.
__global__ __launch_bounds__(256)
void k_gemm1(const float* __restrict__ u, const float* __restrict__ pW,
             const float* __restrict__ pb, bf16* __restrict__ hatz) {
  __shared__ __align__(16) bf16 As[2][128 * 32];
  __shared__ __align__(16) bf16 Bt[2][128 * 32];
  const int tid = threadIdx.x, lane = tid & 31, w = tid >> 5;
  const int wm = w >> 1, wn = w & 1;
  const int m0 = blockIdx.y * 128, n0 = blockIdx.x * 128;
  const int ar = tid >> 1, ah = (tid & 1) * 16;     // A staging: row, k-half
  const int bk = tid >> 3, bn = (tid & 7) * 16;     // B staging: k, n-group

  const float* aSrc = u + (size_t)(m0 + ar) * DIMC + ah;
  float aReg[16], bReg[16];

  // tile 0 -> regs -> LDS[0]
#pragma unroll
  for (int i = 0; i < 16; ++i) aReg[i] = aSrc[i];
  {
    const float* bs = pW + (size_t)bk * NPROJ + n0 + bn;
#pragma unroll
    for (int i = 0; i < 16; ++i) bReg[i] = bs[i];
  }
  {
    v8bf p0, p1;
#pragma unroll
    for (int i = 0; i < 8; ++i) { p0[i] = (bf16)aReg[i]; p1[i] = (bf16)aReg[i + 8]; }
    bf16* ad = As[0] + ar * 32 + ah;
    *(v8bf*)ad = p0; *(v8bf*)(ad + 8) = p1;
#pragma unroll
    for (int i = 0; i < 16; ++i) Bt[0][(bn + i) * 32 + bk] = (bf16)bReg[i];
  }
  __syncthreads();

  v8f acc[2][4] = {};
  const int NIT = DIMC / 32;   // 16
  for (int it = 0; it < NIT; ++it) {
    const int cur = it & 1, nxt = cur ^ 1;
    if (it + 1 < NIT) {
      const int k0 = (it + 1) * 32;
      const float* as = aSrc + k0;
      const float* bs = pW + (size_t)(k0 + bk) * NPROJ + n0 + bn;
      __builtin_prefetch(as + 32, 0, 1);                     // tile after next
      __builtin_prefetch(bs + (size_t)32 * NPROJ, 0, 1);
#pragma unroll
      for (int i = 0; i < 16; ++i) aReg[i] = as[i];
#pragma unroll
      for (int i = 0; i < 16; ++i) bReg[i] = bs[i];
    }
    v16bf a[2], b[4];
#pragma unroll
    for (int tm = 0; tm < 2; ++tm) a[tm] = frag_a(As[cur], wm * 32 + tm * 16, lane);
#pragma unroll
    for (int tn = 0; tn < 4; ++tn) b[tn] = frag_b(Bt[cur], wn * 64 + tn * 16, lane);
#pragma unroll
    for (int tm = 0; tm < 2; ++tm)
#pragma unroll
      for (int tn = 0; tn < 4; ++tn)
        acc[tm][tn] = __builtin_amdgcn_wmma_f32_16x16x32_bf16(
            false, a[tm], false, b[tn], (short)0, acc[tm][tn], false, false);
    if (it + 1 < NIT) {
      v8bf p0, p1;
#pragma unroll
      for (int i = 0; i < 8; ++i) { p0[i] = (bf16)aReg[i]; p1[i] = (bf16)aReg[i + 8]; }
      bf16* ad = As[nxt] + ar * 32 + ah;
      *(v8bf*)ad = p0; *(v8bf*)(ad + 8) = p1;
#pragma unroll
      for (int i = 0; i < 16; ++i) Bt[nxt][(bn + i) * 32 + bk] = (bf16)bReg[i];
    }
    __syncthreads();
  }
#pragma unroll
  for (int tm = 0; tm < 2; ++tm)
#pragma unroll
    for (int tn = 0; tn < 4; ++tn) {
      int n = n0 + wn * 64 + tn * 16 + (lane & 15);
      float pbv = pb[n];
#pragma unroll
      for (int r = 0; r < 8; ++r) {
        int m = m0 + wm * 32 + tm * 16 + r + (lane >> 4) * 8;
        hatz[(size_t)m * NPROJ + n] = (bf16)(acc[tm][tn][r] + pbv);
      }
    }
}

// ---------------- GEMM2: z = conv1d(hat_z, cW) + cb, written transposed -----
// z[m,o] = cb[o] + sum_tap sum_i hatz[m+tap-1,i]*Wt[tap][i][o], edge-masked.
// Flattened, double-buffered pipeline over 3*64 = 192 K-steps.
__global__ __launch_bounds__(256)
void k_gemm2(const bf16* __restrict__ hatz, const bf16* __restrict__ Wt,
             const float* __restrict__ cb, float* __restrict__ zT) {
  __shared__ __align__(16) bf16 As[2][128 * 32];
  __shared__ __align__(16) bf16 Bt[2][128 * 32];
  const int tid = threadIdx.x, lane = tid & 31, w = tid >> 5;
  const int wm = w >> 1, wn = w & 1;
  const int m0 = blockIdx.y * 128, n0 = blockIdx.x * 128;
  const int ar = tid >> 1, ah = (tid & 1) * 16;
  const int bk = tid >> 3, bn = (tid & 7) * 16;

  auto loadA = [&](int tap, int k0, v8bf& a0, v8bf& a1) {
    int ls = ((m0 + ar) & (SEQ - 1)) + tap - 1;
    if ((unsigned)ls < (unsigned)SEQ) {
      const bf16* src = hatz + (size_t)(m0 + ar + tap - 1) * NPROJ + k0 + ah;
      a0 = *(const v8bf*)src; a1 = *(const v8bf*)(src + 8);
    } else {
#pragma unroll
      for (int i = 0; i < 8; ++i) { a0[i] = (bf16)0.f; a1[i] = (bf16)0.f; }
    }
  };
  auto loadB = [&](int tap, int k0, v8bf& b0, v8bf& b1) {
    const bf16* src = Wt + (size_t)tap * NPROJ * NPROJ + (size_t)(k0 + bk) * NPROJ + n0 + bn;
    b0 = *(const v8bf*)src; b1 = *(const v8bf*)(src + 8);
  };
  auto stage = [&](bf16* Ad, bf16* Btd, v8bf a0, v8bf a1, v8bf b0, v8bf b1) {
    bf16* ad = Ad + ar * 32 + ah;
    *(v8bf*)ad = a0; *(v8bf*)(ad + 8) = a1;
#pragma unroll
    for (int i = 0; i < 8; ++i) {
      Btd[(bn + i) * 32 + bk]     = b0[i];
      Btd[(bn + 8 + i) * 32 + bk] = b1[i];
    }
  };

  v8bf a0, a1, b0, b1;
  loadA(0, 0, a0, a1);
  loadB(0, 0, b0, b1);
  stage(As[0], Bt[0], a0, a1, b0, b1);
  __syncthreads();

  v8f acc[2][4] = {};
  const int NIT = 3 * (NPROJ / 32);   // 192
  for (int it = 0; it < NIT; ++it) {
    const int cur = it & 1, nxt = cur ^ 1;
    if (it + 1 < NIT) {
      const int tap = (it + 1) >> 6;
      const int k0  = ((it + 1) & 63) * 32;
      __builtin_prefetch(hatz + (size_t)(m0 + ar) * NPROJ + k0 + 32, 0, 1);
      __builtin_prefetch(Wt + (size_t)tap * NPROJ * NPROJ + (size_t)(k0 + 32 + bk) * NPROJ + n0, 0, 1);
      loadA(tap, k0, a0, a1);
      loadB(tap, k0, b0, b1);
    }
    v16bf a[2], b[4];
#pragma unroll
    for (int tm = 0; tm < 2; ++tm) a[tm] = frag_a(As[cur], wm * 32 + tm * 16, lane);
#pragma unroll
    for (int tn = 0; tn < 4; ++tn) b[tn] = frag_b(Bt[cur], wn * 64 + tn * 16, lane);
#pragma unroll
    for (int tm = 0; tm < 2; ++tm)
#pragma unroll
      for (int tn = 0; tn < 4; ++tn)
        acc[tm][tn] = __builtin_amdgcn_wmma_f32_16x16x32_bf16(
            false, a[tm], false, b[tn], (short)0, acc[tm][tn], false, false);
    if (it + 1 < NIT) stage(As[nxt], Bt[nxt], a0, a1, b0, b1);
    __syncthreads();
  }
#pragma unroll
  for (int tm = 0; tm < 2; ++tm)
#pragma unroll
    for (int tn = 0; tn < 4; ++tn) {
      int n = n0 + wn * 64 + tn * 16 + (lane & 15);
      float cbv = cb[n];
#pragma unroll
      for (int r = 0; r < 8; ++r) {
        int m = m0 + wm * 32 + tm * 16 + r + (lane >> 4) * 8;
        int b_ = m >> 12, l = m & (SEQ - 1);
        zT[((size_t)(b_ * NPROJ + n)) * SEQ + l] = acc[tm][tn][r] + cbv;
      }
    }
}

// ---------------- cW (O,I,3) f32 -> Wt[tap][i][o] bf16 -----------------------
__global__ void k_prep_w(const float* __restrict__ cW, bf16* __restrict__ Wt) {
  size_t idx = (size_t)blockIdx.x * 256 + threadIdx.x; // 3*2048*2048
  int o = (int)(idx & 2047);
  int i = (int)((idx >> 11) & 2047);
  int tap = (int)(idx >> 22);
  Wt[idx] = (bf16)cW[((size_t)o * NPROJ + i) * 3 + tap];
}

// ---------------- in-LDS radix-2 FFT, length 4096, LDS twiddle table --------
#define TWO_PI_N (1.5339807878856412e-3f)   // 2*pi/4096
// tw[e] = exp(-i*2*pi*e/4096), e in [0,2048)
__device__ __forceinline__ void build_tw(float* twr, float* twi, int tid) {
  for (int e = tid; e < 2048; e += 256) {
    float sn, cs;
    __sincosf(-TWO_PI_N * (float)e, &sn, &cs);
    twr[e] = cs; twi[e] = sn;
  }
}
__device__ __forceinline__ void fft_dif(float* re, float* im,
                                        const float* twr, const float* twi, int tid) {
#pragma unroll 1
  for (int s = 0; s < 12; ++s) {
    int len = 2048 >> s, sh = 11 - s;
    __syncthreads();
    for (int t = tid; t < 2048; t += 256) {
      int j = t & (len - 1), g = t >> sh;
      int base = (g << (sh + 1)) + j, p = base + len;
      int e = j << s;
      float cs = twr[e], sn = twi[e];
      float ar = re[base], ai = im[base], br = re[p], bi = im[p];
      re[base] = ar + br; im[base] = ai + bi;
      float dr = ar - br, di = ai - bi;
      re[p] = dr * cs - di * sn;
      im[p] = dr * sn + di * cs;
    }
  }
}
__device__ __forceinline__ void fft_dit_inv(float* re, float* im,
                                            const float* twr, const float* twi, int tid) {
#pragma unroll 1
  for (int s = 0; s < 12; ++s) {
    int len = 1 << s;
    __syncthreads();
    for (int t = tid; t < 2048; t += 256) {
      int j = t & (len - 1), g = t >> s;
      int base = (g << (s + 1)) + j, p = base + len;
      int e = j << (11 - s);
      float cs = twr[e], sn = -twi[e];          // conjugate
      float br = re[p], bi = im[p];
      float tr = br * cs - bi * sn, ti = br * sn + bi * cs;
      float ar = re[base], ai = im[base];
      re[base] = ar + tr; im[base] = ai + ti;
      re[p] = ar - tr;    im[p] = ai - ti;
    }
  }
}

// ---------------- filter FFT: h = (t @ fW + fb)*hann, per channel -----------
__global__ __launch_bounds__(256)
void k_fft_filter(const float* __restrict__ fW, const float* __restrict__ fb,
                  float* __restrict__ HfRe, float* __restrict__ HfIm) {
  __shared__ float re[SEQ];
  __shared__ float im[SEQ];
  __shared__ float twr[2048], twi[2048];
  __shared__ float fwc[HIDC];
  const int tid = threadIdx.x, ch = blockIdx.x; // 0..1535
  for (int j = tid; j < HIDC; j += 256) fwc[j] = fW[(size_t)j * (ORD * DIMC) + ch];
  build_tw(twr, twi, tid);
  __syncthreads();
  const float fbv = fb[ch];
  const float ln1e4 = 9.210340371976184f;
  for (int l = tid; l < SEQ; l += 256) {
    float acc = fbv;
#pragma unroll 4
    for (int i = 0; i < HIDC / 2; ++i) {
      float dv = __expf(-ln1e4 * (2.0f * i) / (float)HIDC);
      float sn, cs;
      __sincosf((float)l * dv, &sn, &cs);
      acc += sn * fwc[2 * i] + cs * fwc[2 * i + 1];
    }
    float win = 0.5f * (1.0f - __cosf(6.283185307179586f * (float)l / (float)(SEQ - 1)));
    re[l] = acc * win;
    im[l] = 0.f;
  }
  fft_dif(re, im, twr, twi, tid);
  __syncthreads();
  for (int i = tid; i < SEQ; i += 256) {
    HfRe[(size_t)ch * SEQ + i] = re[i];
    HfIm[(size_t)ch * SEQ + i] = im[i];
  }
}

// ---------------- one Hyena order: v = x_k * circconv(h_k, v), in place -----
__global__ __launch_bounds__(256)
void k_fft_conv(float* __restrict__ zT, const float* __restrict__ HfRe,
                const float* __restrict__ HfIm, int k) {
  __shared__ float re[SEQ];
  __shared__ float im[SEQ];
  __shared__ float twr[2048], twi[2048];
  const int tid = threadIdx.x, bc = blockIdx.x;
  const int b = bc >> 9, c = bc & (DIMC - 1);
  float* vrow = zT + ((size_t)(b * NPROJ + c)) * SEQ;
  const float* xrow = zT + ((size_t)(b * NPROJ + DIMC + k * DIMC + c)) * SEQ;
  const float* hr = HfRe + (size_t)(k * DIMC + c) * SEQ;
  const float* hi = HfIm + (size_t)(k * DIMC + c) * SEQ;
  build_tw(twr, twi, tid);
  for (int i = tid; i < SEQ; i += 256) { re[i] = vrow[i]; im[i] = 0.f; }
  fft_dif(re, im, twr, twi, tid);
  __syncthreads();
  for (int i = tid; i < SEQ; i += 256) {
    float vr = re[i], vi = im[i], fr = hr[i], fi = hi[i];
    re[i] = vr * fr - vi * fi;
    im[i] = vr * fi + vi * fr;
  }
  fft_dit_inv(re, im, twr, twi, tid);
  __syncthreads();
  const float invN = 1.0f / (float)SEQ;
  for (int i = tid; i < SEQ; i += 256)
    vrow[i] = xrow[i] * (re[i] * invN);
}

// ---------------- out = v @ oW + ob (v read from transposed zT) -------------
__global__ __launch_bounds__(256)
void k_out(const float* __restrict__ zT, const float* __restrict__ oW,
           const float* __restrict__ ob, float* __restrict__ out) {
  int t = blockIdx.x * 256 + threadIdx.x;
  int j = t & 31;
  int l = (t >> 5) & (SEQ - 1);
  int b = t >> 17;
  float acc = ob[j];
  const float* vp = zT + ((size_t)b * NPROJ) * SEQ + l;
#pragma unroll 4
  for (int c = 0; c < DIMC; ++c)
    acc += vp[(size_t)c * SEQ] * oW[c * OUTC + j];
  out[t] = acc;
}

extern "C" void kernel_launch(void* const* d_in, const int* in_sizes, int n_in,
                              void* d_out, int out_size, void* d_ws, size_t ws_size,
                              hipStream_t stream) {
  const float* u  = (const float*)d_in[0];
  const float* pW = (const float*)d_in[1];
  const float* pb = (const float*)d_in[2];
  const float* cW = (const float*)d_in[3];
  const float* cb = (const float*)d_in[4];
  const float* fW = (const float*)d_in[5];
  const float* fb = (const float*)d_in[6];
  const float* oW = (const float*)d_in[7];
  const float* ob = (const float*)d_in[8];
  float* out = (float*)d_out;

  char* ws = (char*)d_ws;
  bf16*  hatz = (bf16*)(ws);                           // 16384*2048 bf16 = 64 MB
  bf16*  Wt   = (bf16*)(ws + (size_t)67108864);        // 3*2048*2048 bf16 = 24 MB
  float* zT   = (float*)(ws + (size_t)92274688);       // 4*2048*4096 f32 = 128 MB
  float* HfRe = (float*)(ws + (size_t)226492416);      // 1536*4096 f32 = 24 MB
  float* HfIm = (float*)(ws + (size_t)251658240);      // 24 MB (end ~264 MB)

  k_prep_w<<<(3 * NPROJ * NPROJ) / 256, 256, 0, stream>>>(cW, Wt);
  k_gemm1<<<dim3(NPROJ / 128, (BATCH * SEQ) / 128), 256, 0, stream>>>(u, pW, pb, hatz);
  k_gemm2<<<dim3(NPROJ / 128, (BATCH * SEQ) / 128), 256, 0, stream>>>(hatz, Wt, cb, zT);
  k_fft_filter<<<ORD * DIMC, 256, 0, stream>>>(fW, fb, HfRe, HfIm);
  for (int k = 0; k < ORD; ++k)
    k_fft_conv<<<BATCH * DIMC, 256, 0, stream>>>(zT, HfRe, HfIm, k);
  k_out<<<(BATCH * SEQ * OUTC) / 256, 256, 0, stream>>>(zT, oW, ob, out);
}
